// PolarBEVTransform_29935922053386
// MI455X (gfx1250) — matrix operand order repaired and checked by reading, not compiled, not run on hardware
//
#include <hip/hip_runtime.h>
#include <hip/hip_bf16.h>

// ---------------- problem constants ----------------
#define IH 256
#define IW 704
#define FH 32
#define FW 88
#define PIX (FH*FW)          // 2816 pixels per image
#define NIMG 2               // B*N
#define CIN 256
#define HID 64
#define D_CH 118             // depth bins
#define COUT 80
#define OCH (D_CH + COUT)    // 198
#define N_R 128
#define N_THETA 256
#define BN_EPS 1e-5f

typedef __attribute__((ext_vector_type(16))) __bf16 v16bf;
typedef __attribute__((ext_vector_type(8)))  float  v8f;

// Native f32 -> bf16 convert: clang lowers fptrunc to the gfx1250 hardware
// convert (packs pairs into v_cvt_pk_bf16_f32 where possible), replacing the
// 3-VALU-op manual RNE sequence from the previous round.
__device__ __forceinline__ __bf16 f2bf(float f) { return (__bf16)f; }

// ============================================================
// Kernel 1: fused  h2 = BN2(W2 @ relu(BN1(W1 @ x)))  + softmax
// One wave = one 16-pixel tile.  GEMMs run as W(MxK) x X(K x pixels)
// so the pixel dimension (contiguous in memory) is the WMMA N dim.
// Outputs written channel-major: depth_ws[b][o][p], feat_ws[b][c][p]
// ============================================================
__global__ __launch_bounds__(256)
void mlp_softmax_kernel(const float* __restrict__ x,
                        const float* __restrict__ w1, const float* __restrict__ b1,
                        const float* __restrict__ g1, const float* __restrict__ be1,
                        const float* __restrict__ m1, const float* __restrict__ v1,
                        const float* __restrict__ w2, const float* __restrict__ b2,
                        const float* __restrict__ g2, const float* __restrict__ be2,
                        const float* __restrict__ m2, const float* __restrict__ v2,
                        float* __restrict__ depth_ws, float* __restrict__ feat_ws)
{
  __shared__ unsigned short h1s[8][16 * HID];   // per-wave 16px x 64ch bf16 bounce (16 KB)

  const int tid  = threadIdx.x;
  const int wave = tid >> 5;
  const int lane = tid & 31;
  const int tile = blockIdx.x * 8 + wave;       // 352 tiles total (44 blocks x 8 waves)
  const int img  = tile / (PIX / 16);           // image index 0..1
  const int p0   = (tile % (PIX / 16)) * 16;    // first pixel of tile
  const int pl   = lane & 15;                   // pixel-in-tile (WMMA N)
  const int hi   = lane >> 4;                   // lane half
  const int p    = p0 + pl;

  const float* X = x + (size_t)img * CIN * PIX;

  // ---- B fragments for GEMM1: X tile, K=256 in 8 chunks of 32 ----
  // B k order used here (contiguous, same as A): lanes 0-15 hold K=0..15, lanes 16-31 K=16..31,
  // element e <-> k_local = e (+16 for upper half)
  v16bf Bf[8];
  for (int kc = 0; kc < 8; ++kc) {
    v16bf f;
    const int kb = kc * 32 + hi * 16;
#pragma unroll
    for (int e = 0; e < 16; ++e)
      f[e] = f2bf(X[(size_t)(kb + e) * PIX + p]);
    Bf[kc] = f;
  }

  // ---- GEMM1: out[o][p] = sum_k W1[o][k] * X[k][p], 4 M-tiles ----
  for (int mt = 0; mt < 4; ++mt) {
    v8f acc = {};
    const int row = mt * 16 + pl;               // A-matrix row (output channel)
#pragma unroll
    for (int kc = 0; kc < 8; ++kc) {
      v16bf Af;
#pragma unroll
      for (int e = 0; e < 16; ++e) {
        const int vv = e >> 1;
        const int k  = kc * 32 + hi * 16 + e; (void)vv;
        Af[e] = f2bf(w1[row * CIN + k]);
      }
      acc = __builtin_amdgcn_wmma_f32_16x16x32_bf16(false, Af, false, Bf[kc],
                                                    (short)0, acc, false, false);
    }
    // BN1 + ReLU, stage to LDS in [pixel][channel] bf16
#pragma unroll
    for (int r = 0; r < 8; ++r) {
      const int o  = mt * 16 + r + hi * 8;      // channel of acc[r] (D layout)
      const float inv = g1[o] * rsqrtf(v1[o] + BN_EPS);
      float y = (acc[r] + b1[o]) * inv + (be1[o] - m1[o] * inv);
      y = fmaxf(y, 0.0f);
      h1s[wave][pl * HID + o] = __builtin_bit_cast(unsigned short, f2bf(y));
    }
  }
  __syncthreads();

  // ---- B fragments for GEMM2 from LDS h1 (K=64, 2 chunks) ----
  v16bf B2[2];
  for (int kc = 0; kc < 2; ++kc) {
    v16bf f;
    const int kb = kc * 32 + hi * 16;
#pragma unroll
    for (int e = 0; e < 16; ++e)
      f[e] = __builtin_bit_cast(__bf16, h1s[wave][pl * HID + kb + e]);
    B2[kc] = f;
  }

  // ---- GEMM2: 13 M-tiles (198 channels, zero-padded to 208) + BN2 ----
  float h2[104];
  for (int mt = 0; mt < 13; ++mt) {
    v8f acc = {};
    const int row = mt * 16 + pl;
#pragma unroll
    for (int kc = 0; kc < 2; ++kc) {
      v16bf Af;
#pragma unroll
      for (int e = 0; e < 16; ++e) {
        const int vv = e >> 1;
        const int k  = kc * 32 + hi * 16 + e; (void)vv;
        Af[e] = (row < OCH) ? f2bf(w2[row * HID + k]) : f2bf(0.0f);
      }
      acc = __builtin_amdgcn_wmma_f32_16x16x32_bf16(false, Af, false, B2[kc],
                                                    (short)0, acc, false, false);
    }
#pragma unroll
    for (int r = 0; r < 8; ++r) {
      const int o = mt * 16 + r + hi * 8;
      float y = 0.0f;
      if (o < OCH) {
        const float inv = g2[o] * rsqrtf(v2[o] + BN_EPS);
        y = (acc[r] + b2[o]) * inv + (be2[o] - m2[o] * inv);
      }
      h2[mt * 8 + r] = y;
    }
  }

  // ---- softmax over depth channels (o < 118); each pixel's channels live
  //      in lanes pl and pl+16 -> combine with shfl_xor(16) on wave32 ----
  float mx = -3.0e38f;
#pragma unroll
  for (int mt = 0; mt < 13; ++mt)
#pragma unroll
    for (int r = 0; r < 8; ++r) {
      const int o = mt * 16 + r + hi * 8;
      if (o < D_CH) mx = fmaxf(mx, h2[mt * 8 + r]);
    }
  mx = fmaxf(mx, __shfl_xor(mx, 16, 32));

  float s = 0.0f;
#pragma unroll
  for (int mt = 0; mt < 13; ++mt)
#pragma unroll
    for (int r = 0; r < 8; ++r) {
      const int o = mt * 16 + r + hi * 8;
      if (o < D_CH) s += __expf(h2[mt * 8 + r] - mx);
    }
  s += __shfl_xor(s, 16, 32);
  const float rs = 1.0f / s;

  // ---- write channel-major outputs (coalesced-in-pixel for kernel 2) ----
#pragma unroll
  for (int mt = 0; mt < 13; ++mt)
#pragma unroll
    for (int r = 0; r < 8; ++r) {
      const int o = mt * 16 + r + hi * 8;
      const float y = h2[mt * 8 + r];
      if (o < D_CH)
        depth_ws[((size_t)img * D_CH + o) * PIX + p] = __expf(y - mx) * rs;
      else if (o < OCH)
        feat_ws[((size_t)img * COUT + (o - D_CH)) * PIX + p] = y;
    }
}

// ============================================================
// Kernel 2: geometry + lift + scatter-add into polar BEV.
// One thread per (b, d, pixel); 80 coalesced feat loads + 80
// global_atomic_add_f32 into the L2-resident 21MB output.
// ============================================================
__device__ __forceinline__ int upper_bound_m1(const float* e, int n, float xv) {
  int lo = 0, high = n;                 // first index with e[i] > xv
  while (lo < high) {
    const int mid = (lo + high) >> 1;
    if (e[mid] <= xv) lo = mid + 1; else high = mid;
  }
  return lo - 1;                        // searchsorted(..., 'right') - 1
}

__global__ __launch_bounds__(256)
void lift_scatter_kernel(const float* __restrict__ depth_ws,
                         const float* __restrict__ feat_ws,
                         const float* __restrict__ rots,
                         const float* __restrict__ trans,
                         const float* __restrict__ intr,
                         float* __restrict__ out)
{
  __shared__ float rad_e[N_R + 1];      // nonuniform radius edges
  __shared__ float ang_e[N_THETA + 1];  // uniform angle edges

  const int tid = threadIdx.x;
  for (int i = tid; i < N_R + 1; i += blockDim.x) {
    const float t = (float)i / (float)N_R;
    rad_e[i] = 1.0f + powf(t, 1.5f) * 59.0f;
  }
  const float PIf = 3.14159265358979323846f;
  for (int i = tid; i < N_THETA + 1; i += blockDim.x)
    ang_e[i] = -0.5f * PIf + (float)i * (PIf / (float)N_THETA);
  __syncthreads();

  const long long idx = (long long)blockIdx.x * blockDim.x + tid; // b*D*PIX items
  const int p = (int)(idx % PIX);
  const long long t1 = idx / PIX;
  const int d = (int)(t1 % D_CH);
  const int b = (int)(t1 / D_CH);

  // frustum point
  const int h = p / FW, w = p % FW;
  const float dep = 1.0f + 0.5f * (float)d;
  const float u  = (float)w * ((float)(IW - 1) / (float)(FW - 1));
  const float vv = (float)h * ((float)(IH - 1) / (float)(FH - 1));

  const float* K = intr + b * 9;
  const float fx = K[0], cx = K[2], fy = K[4], cy = K[5];
  const float pcx = (u - cx) * dep / fx;
  const float pcy = (vv - cy) * dep / fy;
  const float pcz = dep;

  const float* R = rots + b * 9;
  const float* T = trans + b * 3;
  const float xl = R[0] * pcx + R[1] * pcy + R[2] * pcz + T[0];
  const float yl = R[3] * pcx + R[4] * pcy + R[5] * pcz + T[1];

  const float r  = sqrtf(xl * xl + yl * yl);
  const float th = atan2f(yl, xl);

  const int ri = upper_bound_m1(rad_e, N_R + 1, r);
  const int ti = upper_bound_m1(ang_e, N_THETA + 1, th);
  if (ri < 0 || ri >= N_R || ti < 0 || ti >= N_THETA) return;

  const float dval = depth_ws[((size_t)b * D_CH + d) * PIX + p];
  const float* fptr = feat_ws + (size_t)b * COUT * PIX + p;
  float* obase = out + (size_t)b * COUT * N_R * N_THETA
                     + (size_t)ri * N_THETA + ti;

#pragma unroll 4
  for (int c = 0; c < COUT; ++c) {
    const float val = dval * fptr[(size_t)c * PIX];
    (void)__hip_atomic_fetch_add(obase + (size_t)c * (N_R * N_THETA), val,
                                 __ATOMIC_RELAXED, __HIP_MEMORY_SCOPE_AGENT);
  }
}

// ============================================================
extern "C" void kernel_launch(void* const* d_in, const int* in_sizes, int n_in,
                              void* d_out, int out_size, void* d_ws, size_t ws_size,
                              hipStream_t stream) {
  const float* x     = (const float*)d_in[0];
  const float* rots  = (const float*)d_in[1];
  const float* trans = (const float*)d_in[2];
  const float* intr  = (const float*)d_in[3];
  const float* w1  = (const float*)d_in[4];
  const float* b1  = (const float*)d_in[5];
  const float* g1  = (const float*)d_in[6];
  const float* be1 = (const float*)d_in[7];
  const float* m1  = (const float*)d_in[8];
  const float* v1  = (const float*)d_in[9];
  const float* w2  = (const float*)d_in[10];
  const float* b2  = (const float*)d_in[11];
  const float* g2  = (const float*)d_in[12];
  const float* be2 = (const float*)d_in[13];
  const float* m2  = (const float*)d_in[14];
  const float* v2  = (const float*)d_in[15];

  float* depth_ws = (float*)d_ws;                              // [2][118][2816]
  float* feat_ws  = depth_ws + (size_t)NIMG * D_CH * PIX;      // [2][80][2816]

  // Kernel 1: 352 wave-tiles / 8 waves per block = 44 blocks
  mlp_softmax_kernel<<<44, 256, 0, stream>>>(
      x, w1, b1, g1, be1, m1, v1, w2, b2, g2, be2, m2, v2, depth_ws, feat_ws);

  // zero the BEV accumulator (graph-capture-legal async memset)
  hipMemsetAsync(d_out, 0, (size_t)out_size * sizeof(float), stream);

  // Kernel 2: 2*118*2816 = 664576 items, 256 threads/block = 2596 blocks
  lift_scatter_kernel<<<2596, 256, 0, stream>>>(
      depth_ws, feat_ws, rots, trans, intr, (float*)d_out);
}
